// DGCNN_61134564491270
// MI455X (gfx1250) — matrix operand order repaired and checked by reading, not compiled, not run on hardware
//
#include <hip/hip_runtime.h>

#define N_PTS 1024
#define BATCH 16
#define KNN   20
#define NEG_SLOPE 0.2f
#define NEG_INF  -3.4e38f

typedef __attribute__((ext_vector_type(16))) __bf16 v16bf;
typedef __attribute__((ext_vector_type(8)))  float  v8f;
typedef __attribute__((ext_vector_type(8)))  unsigned v8u;

union BF { v16bf bf; v8u u; };

// ---------------------------------------------------------------------------
// WMMA helper: D = A(16x32 bf16) * B(32x16 bf16) + C(16x16 f32)
// ---------------------------------------------------------------------------
__device__ __forceinline__ v8f wmma_bf16(v16bf a, v16bf b, v8f c) {
  return __builtin_amdgcn_wmma_f32_16x16x32_bf16(
      /*neg_a=*/false, a, /*neg_b=*/false, b,
      /*c_mod=*/(short)0, c, /*reuse_a=*/false, /*reuse_b=*/false);
}

// packed bf16 pair subtract: a - b via sign-flip + v_pk_add_bf16 (VOP3P)
__device__ __forceinline__ unsigned pk_sub_bf16(unsigned a, unsigned b) {
  unsigned d;
  asm("v_pk_add_bf16 %0, %1, %2" : "=v"(d) : "v"(a), "v"(b ^ 0x80008000u));
  return d;
}

__device__ __forceinline__ unsigned pack_bf(float a, float b) {
  union { unsigned u; __bf16 h[2]; } c;
  c.h[0] = (__bf16)a;
  c.h[1] = (__bf16)b;
  return c.u;
}
__device__ __forceinline__ float fval_pk(const unsigned* f, int c, int n) {
  union { unsigned u; __bf16 h[2]; } cc;
  cc.u = f[(long)(c >> 1) * N_PTS + n];
  return (float)cc.h[c & 1];
}

// A-fragment 16x32 bf16 (contiguous k = 16h + 2v, matches B): channel-pair index per dword slot v
__device__ __forceinline__ int a_kbase(int v, int h) {
  return h * 16 + 2 * v;
}
__device__ __forceinline__ int b_kbase(int v, int h) { return h * 16 + 2 * v; }

// A-frag from f32 weight matrix W[O][CI] (one-time, hoisted to registers)
__device__ __forceinline__ v16bf load_a_w(const float* __restrict__ W, int CI,
                                          int o, int c0, int h) {
  v16bf a;
#pragma unroll
  for (int v = 0; v < 8; ++v) {
    int kb = c0 + a_kbase(v, h);
    float x0 = (kb     < CI) ? W[(long)o * CI + kb]     : 0.f;
    float x1 = (kb + 1 < CI) ? W[(long)o * CI + kb + 1] : 0.f;
    a[2 * v]     = (__bf16)x0;
    a[2 * v + 1] = (__bf16)x1;
  }
  return a;
}

// A-frag from packed bf16 feature [P][N], row = point i, K = channel
template <int P>
__device__ __forceinline__ v16bf load_a_pk(const unsigned* __restrict__ f,
                                           int i, int c0, int h) {
  BF r;
#pragma unroll
  for (int v = 0; v < 8; ++v) {
    int kp = (c0 + a_kbase(v, h)) >> 1;
    r.u[v] = (kp < P) ? f[(long)kp * N_PTS + i] : 0u;
  }
  return r.bf;
}

// B-frag from packed bf16 feature [P][N], K = channel, col = point n
template <int P>
__device__ __forceinline__ v16bf load_b_pk(const unsigned* __restrict__ f,
                                           int n, int c0, int h) {
  BF r;
#pragma unroll
  for (int v = 0; v < 8; ++v) {
    int kp = (c0 >> 1) + h * 8 + v;
    r.u[v] = (kp < P) ? f[(long)kp * N_PTS + n] : 0u;
  }
  return r.bf;
}

// scalar edge-feature frag (small-C path; C=3 so diff/center split a pair)
__device__ __forceinline__ v16bf load_b_edge_small(const unsigned* __restrict__ f,
                                                   int C, int ctr, int nbr,
                                                   int h) {
  v16bf b;
#pragma unroll
  for (int v = 0; v < 8; ++v) {
#pragma unroll
    for (int s = 0; s < 2; ++s) {
      int kc = b_kbase(v, h) + s;
      float val = 0.f;
      if (kc < C)
        val = fval_pk(f, kc, nbr) - fval_pk(f, kc, ctr);
      else if (kc < 2 * C)
        val = fval_pk(f, kc - C, ctr);
      b[2 * v + s] = (__bf16)val;
    }
  }
  return b;
}

__device__ __forceinline__ float lrelu(float v) {
  return (v >= 0.f) ? v : NEG_SLOPE * v;
}

// ---------------------------------------------------------------------------
// f32 [C][N] -> packed bf16 pairs [P][N]  (only needed for the raw input x)
// ---------------------------------------------------------------------------
__global__ void k_cvt_pk(const float* __restrict__ f, long bsf, int C,
                         unsigned* __restrict__ o, long bso) {
  int b = blockIdx.z, p = blockIdx.y;
  int n = blockIdx.x * blockDim.x + threadIdx.x;
  const float* fb = f + (long)b * bsf;
  float x0 = fb[(long)(2 * p) * N_PTS + n];
  float x1 = (2 * p + 1 < C) ? fb[(long)(2 * p + 1) * N_PTS + n] : 0.f;
  o[(long)b * bso + (long)p * N_PTS + n] = pack_bf(x0, x1);
}

// squared norms from packed bf16 (consistent with bf16 inner products)
__global__ void k_sqnorm_pk(const unsigned* __restrict__ f, long bstride, int P,
                            float* __restrict__ xx) {
  int b = blockIdx.y;
  int n = blockIdx.x * blockDim.x + threadIdx.x;
  const unsigned* fb = f + (long)b * bstride;
  float s = 0.f;
  for (int p = 0; p < P; ++p) {
    union { unsigned u; __bf16 h[2]; } c;
    c.u = fb[(long)p * N_PTS + n];
    float a = (float)c.h[0], bv = (float)c.h[1];
    s += a * a + bv * bv;
  }
  xx[b * N_PTS + n] = s;
}

// ---------------------------------------------------------------------------
// Fused KNN: block = (b, 16-row strip). 8 waves build a 16x1024 distance
// strip in LDS via WMMA; then 16 threads/row do parallel top-20 selection.
// pd = 2*inner - xx_i - xx_j  (self at ~0 is max -> included, as reference)
// ---------------------------------------------------------------------------
template <int P>
__global__ __launch_bounds__(256) void k_knn(const unsigned* __restrict__ f,
                                             long bstride,
                                             const float* __restrict__ xx,
                                             int* __restrict__ idx) {
  extern __shared__ float dist[];  // [16][N_PTS] = 64 KB dynamic
  __shared__ float redv[256];
  __shared__ int   redi[256];
  constexpr int NCH = (2 * P + 31) / 32;
  int b = blockIdx.y, it = blockIdx.x;
  int i0 = it * 16;
  const unsigned* fb = f + (long)b * bstride;
  const float* xxb = xx + b * N_PTS;
  int tid = threadIdx.x;
  int lane = tid & 31, wid = tid >> 5;
  int h = lane >> 4, m = lane & 15;

  v16bf areg[NCH];
#pragma unroll
  for (int ch = 0; ch < NCH; ++ch)
    areg[ch] = load_a_pk<P>(fb, i0 + m, ch * 32, h);

  for (int jt = wid; jt < N_PTS / 16; jt += 8) {
    int j0 = jt * 16;
    v8f acc = {};
#pragma unroll
    for (int ch = 0; ch < NCH; ++ch) {
      v16bf bb = load_b_pk<P>(fb, j0 + m, ch * 32, h);
      acc = wmma_bf16(areg[ch], bb, acc);
    }
    float xj = xxb[j0 + m];
#pragma unroll
    for (int r = 0; r < 8; ++r) {
      int row = r + 8 * h;
      dist[row * N_PTS + j0 + m] = 2.f * acc[r] - xxb[i0 + row] - xj;
    }
  }
  __syncthreads();

  // parallel top-20: 16 threads per row
  int row = tid >> 4, sub = tid & 15;
  float* drow = dist + row * N_PTS;
  int* op = idx + ((long)b * N_PTS + i0 + row) * KNN;
  for (int t = 0; t < KNN; ++t) {
    float bv = NEG_INF;
    int bi = 0;
    for (int j = sub; j < N_PTS; j += 16) {
      float v = drow[j];
      if (v > bv) { bv = v; bi = j; }
    }
    redv[tid] = bv;
    redi[tid] = bi;
    __syncthreads();
#pragma unroll
    for (int s = 8; s > 0; s >>= 1) {
      if (sub < s && redv[tid + s] > redv[tid]) {
        redv[tid] = redv[tid + s];
        redi[tid] = redi[tid + s];
      }
      __syncthreads();
    }
    if (sub == 0) {
      op[t] = redi[tid];
      drow[redi[tid]] = NEG_INF;
    }
    __syncthreads();
  }
}

// ---------------------------------------------------------------------------
// Fused EdgeConv: out[p_off + o/2][n] = packed max_k lrelu(W . edge(n,k))
// one wave per (b, 16-o tile, 16-n tile). Center-half fragments are
// k-invariant (hoisted); diff-half uses packed bf16 gather + v_pk_add_bf16.
// ---------------------------------------------------------------------------
template <int C>
__global__ __launch_bounds__(32) void k_edgeconv(
    const unsigned* __restrict__ f, long bs_in, const int* __restrict__ idx,
    const float* __restrict__ W, unsigned* __restrict__ out, long bs_out,
    int p_off) {
  constexpr int C2 = 2 * C;
  constexpr int NCH = (C2 + 31) / 32;
  constexpr int P = (C + 1) / 2;
  int b = blockIdx.z, ot = blockIdx.y, nt = blockIdx.x;
  int lane = threadIdx.x & 31;
  int h = lane >> 4, m = lane & 15;
  int o0 = ot * 16, n0 = nt * 16;
  const unsigned* fb = f + (long)b * bs_in;
  const int* ib = idx + (long)b * N_PTS * KNN;
  int ctr = n0 + m;

  v16bf areg[NCH];
#pragma unroll
  for (int ch = 0; ch < NCH; ++ch)
    areg[ch] = load_a_w(W, C2, o0 + m, ch * 32, h);

  v8f best;
#pragma unroll
  for (int r = 0; r < 8; ++r) best[r] = NEG_INF;

  if constexpr (C % 32 == 0) {
    constexpr int NCD = C / 32;  // diff chunks; center chunks = NCD too
    unsigned ctrp[P];
#pragma unroll
    for (int p = 0; p < P; ++p) ctrp[p] = fb[(long)p * N_PTS + ctr];
    BF bctr[NCD];
#pragma unroll
    for (int cc = 0; cc < NCD; ++cc)
#pragma unroll
      for (int v = 0; v < 8; ++v)
        bctr[cc].u[v] = ctrp[cc * 16 + h * 8 + v];

    for (int k = 0; k < KNN; ++k) {
      int nbr = ib[ctr * KNN + k];
      v8f acc = {};
#pragma unroll
      for (int cd = 0; cd < NCD; ++cd) {
        BF r;
#pragma unroll
        for (int v = 0; v < 8; ++v) {
          int kp = cd * 16 + h * 8 + v;
          r.u[v] = pk_sub_bf16(fb[(long)kp * N_PTS + nbr], ctrp[kp]);
        }
        acc = wmma_bf16(areg[cd], r.bf, acc);
      }
#pragma unroll
      for (int cc = 0; cc < NCD; ++cc)
        acc = wmma_bf16(areg[NCD + cc], bctr[cc].bf, acc);
#pragma unroll
      for (int r = 0; r < 8; ++r) best[r] = fmaxf(best[r], lrelu(acc[r]));
    }
  } else {
    for (int k = 0; k < KNN; ++k) {
      int nbr = ib[ctr * KNN + k];
      v16bf bb = load_b_edge_small(fb, C, ctr, nbr, h);
      v8f acc = {};
      acc = wmma_bf16(areg[0], bb, acc);
#pragma unroll
      for (int r = 0; r < 8; ++r) best[r] = fmaxf(best[r], lrelu(acc[r]));
    }
  }

  // write packed bf16 pairs: lane holds rows o0+8h+0..7 -> 4 pairs
  unsigned* ob = out + (long)b * bs_out + (long)p_off * N_PTS;
  int pbase = (o0 + 8 * h) >> 1;
#pragma unroll
  for (int q = 0; q < 4; ++q)
    ob[(long)(pbase + q) * N_PTS + n0 + m] = pack_bf(best[2 * q], best[2 * q + 1]);
}

// ---------------------------------------------------------------------------
// Transform net stage 1+2: h1 = lrelu(tw1 @ edge(x)) [64] staged in LDS
// (packed bf16 pairs); h2 = lrelu(tw2 @ h1) [128]; max over k.
// Block = (b, 16-n tile), 8 waves.
// ---------------------------------------------------------------------------
__global__ __launch_bounds__(256) void k_transform_h12(
    const unsigned* __restrict__ xpk, const int* __restrict__ idx,
    const float* __restrict__ tw1, const float* __restrict__ tw2,
    unsigned* __restrict__ h128pk) {
  __shared__ unsigned h1s[32 * 16];  // [pair][16 cols]
  int b = blockIdx.y, nt = blockIdx.x;
  int tid = threadIdx.x, lane = tid & 31, wid = tid >> 5;
  int h = lane >> 4, m = lane & 15;
  int n0 = nt * 16;
  const unsigned* xb = xpk + (long)b * 2 * N_PTS;
  const int* ib = idx + (long)b * N_PTS * KNN;
  int ctr = n0 + m;

  v16bf a1 = {};
  if (wid < 4) a1 = load_a_w(tw1, 6, wid * 16 + m, 0, h);
  v16bf a2[2];
#pragma unroll
  for (int ch = 0; ch < 2; ++ch)
    a2[ch] = load_a_w(tw2, 64, wid * 16 + m, ch * 32, h);

  v8f best;
#pragma unroll
  for (int r = 0; r < 8; ++r) best[r] = NEG_INF;

  for (int k = 0; k < KNN; ++k) {
    int nbr = ib[ctr * KNN + k];
    if (wid < 4) {  // wave-uniform branch: EXEC stays full inside
      v16bf bb = load_b_edge_small(xb, 3, ctr, nbr, h);
      v8f acc = {};
      acc = wmma_bf16(a1, bb, acc);
      int pbase = (wid * 16 + 8 * h) >> 1;
#pragma unroll
      for (int q = 0; q < 4; ++q)
        h1s[(pbase + q) * 16 + m] =
            pack_bf(lrelu(acc[2 * q]), lrelu(acc[2 * q + 1]));
    }
    __syncthreads();
    v8f acc = {};
#pragma unroll
    for (int ch = 0; ch < 2; ++ch) {
      BF r;
#pragma unroll
      for (int v = 0; v < 8; ++v) r.u[v] = h1s[(ch * 16 + h * 8 + v) * 16 + m];
      acc = wmma_bf16(a2[ch], r.bf, acc);
    }
#pragma unroll
    for (int r = 0; r < 8; ++r) best[r] = fmaxf(best[r], lrelu(acc[r]));
    __syncthreads();
  }

  unsigned* ob = h128pk + (long)b * 64 * N_PTS;
  int pbase = (wid * 16 + 8 * h) >> 1;
#pragma unroll
  for (int q = 0; q < 4; ++q)
    ob[(long)(pbase + q) * N_PTS + n0 + m] = pack_bf(best[2 * q], best[2 * q + 1]);
}

// ---------------------------------------------------------------------------
// out[b][o] = max_n lrelu( W[o][:] . f[b][:][n] )   (tw3 and w5)
// one wave per (b, 16-o tile); packed-input B-frags; shfl_xor column max
// ---------------------------------------------------------------------------
template <int C>
__global__ __launch_bounds__(32) void k_gemm_lrelu_maxn(
    const unsigned* __restrict__ f, long bs_in, const float* __restrict__ W,
    int O, float* __restrict__ out) {
  constexpr int NCH = (C + 31) / 32;
  constexpr int P = C / 2;
  int b = blockIdx.y, ot = blockIdx.x;
  int lane = threadIdx.x & 31;
  int h = lane >> 4, m = lane & 15;
  int o0 = ot * 16;
  const unsigned* fb = f + (long)b * bs_in;

  v16bf areg[NCH];
#pragma unroll
  for (int ch = 0; ch < NCH; ++ch)
    areg[ch] = load_a_w(W, C, o0 + m, ch * 32, h);

  v8f best;
#pragma unroll
  for (int r = 0; r < 8; ++r) best[r] = NEG_INF;

  for (int nt = 0; nt < N_PTS / 16; ++nt) {
    int n = nt * 16 + m;
    v8f acc = {};
#pragma unroll
    for (int ch = 0; ch < NCH; ++ch) {
      v16bf bb = load_b_pk<P>(fb, n, ch * 32, h);
      acc = wmma_bf16(areg[ch], bb, acc);
    }
#pragma unroll
    for (int r = 0; r < 8; ++r) best[r] = fmaxf(best[r], lrelu(acc[r]));
  }
#pragma unroll
  for (int mask = 1; mask < 16; mask <<= 1) {
#pragma unroll
    for (int r = 0; r < 8; ++r)
      best[r] = fmaxf(best[r], __shfl_xor(best[r], mask, 32));
  }
  if (m == 0) {
#pragma unroll
    for (int r = 0; r < 8; ++r)
      out[(long)b * O + o0 + r + 8 * h] = best[r];
  }
}

// ---------------------------------------------------------------------------
// Fused fc layer: l2-normalize -> Linear -> LayerNorm -> LeakyReLU
// ---------------------------------------------------------------------------
__device__ __forceinline__ float block_reduce_sum(float v, float* red, int tid) {
  red[tid] = v;
  __syncthreads();
  for (int s = 128; s > 0; s >>= 1) {
    if (tid < s) red[tid] += red[tid + s];
    __syncthreads();
  }
  float r = red[0];
  __syncthreads();
  return r;
}

template <int CI, int CO, bool HAS_BIAS>
__global__ __launch_bounds__(256) void k_fc(
    const float* __restrict__ in, const float* __restrict__ W,
    const float* __restrict__ bias, const float* __restrict__ g,
    const float* __restrict__ beta, float* __restrict__ out) {
  __shared__ float sin[CI];
  __shared__ float sout[CO];
  __shared__ float red[256];
  int b = blockIdx.x, tid = threadIdx.x;
  const float* ib = in + (long)b * CI;

  float ss = 0.f;
  for (int i = tid; i < CI; i += 256) {
    float v = ib[i];
    sin[i] = v;
    ss += v * v;
  }
  float inv = rsqrtf(block_reduce_sum(ss, red, tid));  // l2n, no eps (ref)

  for (int o = tid; o < CO; o += 256) {
    float acc = 0.f;
    for (int c = 0; c < CI; ++c) acc += W[(long)o * CI + c] * sin[c];
    acc *= inv;
    if (HAS_BIAS) acc += bias[o];
    sout[o] = acc;
  }
  __syncthreads();

  float sm = 0.f, sq = 0.f;
  for (int o = tid; o < CO; o += 256) {
    float v = sout[o];
    sm += v;
    sq += v * v;
  }
  float mean = block_reduce_sum(sm, red, tid) / CO;
  float var = block_reduce_sum(sq, red, tid) / CO - mean * mean;
  float rs = rsqrtf(var + 1e-5f);
  for (int o = tid; o < CO; o += 256) {
    float v = (sout[o] - mean) * rs * g[o] + beta[o];
    out[(long)b * CO + o] = lrelu(v);
  }
}

// t[b] = fc2out[b] @ tfc3_w.T + tfc3_b + eye(3).flatten
__global__ void k_fc3(const float* __restrict__ in, const float* __restrict__ W,
                      const float* __restrict__ bias, float* __restrict__ t) {
  int b = blockIdx.x, o = threadIdx.x;
  if (o < 9) {
    const float* ib = in + (long)b * 256;
    float acc = 0.f;
    for (int c = 0; c < 256; ++c) acc += W[o * 256 + c] * ib[c];
    acc += bias[o];
    if (o == 0 || o == 4 || o == 8) acc += 1.f;
    t[b * 9 + o] = acc;
  }
}

// xt = t (3x3) @ x (3xN), written as packed bf16 pairs [2][N] (ch3 = 0 pad)
__global__ void k_xform(const float* __restrict__ t, const float* __restrict__ x,
                        unsigned* __restrict__ xtpk) {
  int b = blockIdx.y;
  int n = blockIdx.x * blockDim.x + threadIdx.x;
  const float* xb = x + (long)b * 3 * N_PTS;
  const float* tb = t + b * 9;
  float x0 = xb[n], x1 = xb[N_PTS + n], x2 = xb[2 * N_PTS + n];
  float s0 = tb[0] * x0 + tb[1] * x1 + tb[2] * x2;
  float s1 = tb[3] * x0 + tb[4] * x1 + tb[5] * x2;
  float s2 = tb[6] * x0 + tb[7] * x1 + tb[8] * x2;
  unsigned* ob = xtpk + (long)b * 2 * N_PTS;
  ob[n] = pack_bf(s0, s1);
  ob[N_PTS + n] = pack_bf(s2, 0.f);
}

// ---------------------------------------------------------------------------
extern "C" void kernel_launch(void* const* d_in, const int* in_sizes, int n_in,
                              void* d_out, int out_size, void* d_ws,
                              size_t ws_size, hipStream_t stream) {
  const float* x        = (const float*)d_in[0];
  const float* tw1      = (const float*)d_in[1];
  const float* tw2      = (const float*)d_in[2];
  const float* tw3      = (const float*)d_in[3];
  const float* tfc1_w   = (const float*)d_in[4];
  const float* tfc1_g   = (const float*)d_in[5];
  const float* tfc1_b   = (const float*)d_in[6];
  const float* tfc2_w   = (const float*)d_in[7];
  const float* tfc2_bias= (const float*)d_in[8];
  const float* tfc2_g   = (const float*)d_in[9];
  const float* tfc2_b   = (const float*)d_in[10];
  const float* tfc3_w   = (const float*)d_in[11];
  const float* tfc3_b   = (const float*)d_in[12];
  const float* w1       = (const float*)d_in[13];
  const float* w2       = (const float*)d_in[14];
  const float* w3       = (const float*)d_in[15];
  const float* w4       = (const float*)d_in[16];
  const float* w5       = (const float*)d_in[17];
  float* out = (float*)d_out;

  char* ws = (char*)d_ws;
  size_t off = 0;
  auto alloc = [&](size_t bytes) {
    void* p = ws + off;
    off = (off + bytes + 255) & ~(size_t)255;
    return p;
  };
  int*      idx    = (int*)     alloc((size_t)BATCH * N_PTS * KNN * 4);
  float*    xx     = (float*)   alloc((size_t)BATCH * N_PTS * 4);
  unsigned* xpk    = (unsigned*)alloc((size_t)BATCH * 2 * N_PTS * 4);
  unsigned* xtpk   = (unsigned*)alloc((size_t)BATCH * 2 * N_PTS * 4);
  unsigned* h128pk = (unsigned*)alloc((size_t)BATCH * 64 * N_PTS * 4);
  unsigned* xcatpk = (unsigned*)alloc((size_t)BATCH * 256 * N_PTS * 4);
  float*    hglob  = (float*)   alloc((size_t)BATCH * 1024 * 4);
  float*    fc1o   = (float*)   alloc((size_t)BATCH * 512 * 4);
  float*    fc2o   = (float*)   alloc((size_t)BATCH * 256 * 4);
  float*    tmat   = (float*)   alloc((size_t)BATCH * 9 * 4);

  const size_t distLDS = 16 * N_PTS * sizeof(float);  // 64 KB
  dim3 gN(N_PTS / 256, BATCH);
  dim3 gKNN(N_PTS / 16, BATCH);
  const long bs3f = 3L * N_PTS;            // f32 x batch stride
  const long bs2 = 2L * N_PTS;             // packed pair strides
  const long bs64 = 64L * N_PTS;
  const long bs256 = 256L * N_PTS;

  // --- pack raw x; KNN for f0 / transform net ---
  k_cvt_pk<<<dim3(N_PTS / 256, 2, BATCH), 256, 0, stream>>>(x, bs3f, 3, xpk, bs2);
  k_sqnorm_pk<<<gN, 256, 0, stream>>>(xpk, bs2, 2, xx);
  k_knn<2><<<gKNN, 256, distLDS, stream>>>(xpk, bs2, xx, idx);

  // --- transform net ---
  k_transform_h12<<<gKNN, 256, 0, stream>>>(xpk, idx, tw1, tw2, h128pk);
  k_gemm_lrelu_maxn<128><<<dim3(1024 / 16, BATCH), 32, 0, stream>>>(
      h128pk, bs64, tw3, 1024, hglob);
  k_fc<1024, 512, false><<<BATCH, 256, 0, stream>>>(hglob, tfc1_w, nullptr,
                                                    tfc1_g, tfc1_b, fc1o);
  k_fc<512, 256, true><<<BATCH, 256, 0, stream>>>(fc1o, tfc2_w, tfc2_bias,
                                                  tfc2_g, tfc2_b, fc2o);
  k_fc3<<<BATCH, 32, 0, stream>>>(fc2o, tfc3_w, tfc3_b, tmat);
  k_xform<<<gN, 256, 0, stream>>>(tmat, x, xtpk);

  // --- EdgeConv 1: xt (C=3) -> xcat pairs [0,32) ---
  k_sqnorm_pk<<<gN, 256, 0, stream>>>(xtpk, bs2, 2, xx);
  k_knn<2><<<gKNN, 256, distLDS, stream>>>(xtpk, bs2, xx, idx);
  k_edgeconv<3><<<dim3(N_PTS / 16, 64 / 16, BATCH), 32, 0, stream>>>(
      xtpk, bs2, idx, w1, xcatpk, bs256, 0);

  // --- EdgeConv 2: x1 (C=64, pairs 0..31) -> pairs [32,64) ---
  k_sqnorm_pk<<<gN, 256, 0, stream>>>(xcatpk, bs256, 32, xx);
  k_knn<32><<<gKNN, 256, distLDS, stream>>>(xcatpk, bs256, xx, idx);
  k_edgeconv<64><<<dim3(N_PTS / 16, 64 / 16, BATCH), 32, 0, stream>>>(
      xcatpk, bs256, idx, w2, xcatpk, bs256, 32);

  // --- EdgeConv 3: x2 (C=64, pairs 32..63) -> pairs [64,128) ---
  k_sqnorm_pk<<<gN, 256, 0, stream>>>(xcatpk + 32L * N_PTS, bs256, 32, xx);
  k_knn<32><<<gKNN, 256, distLDS, stream>>>(xcatpk + 32L * N_PTS, bs256, xx, idx);
  k_edgeconv<64><<<dim3(N_PTS / 16, 128 / 16, BATCH), 32, 0, stream>>>(
      xcatpk + 32L * N_PTS, bs256, idx, w3, xcatpk, bs256, 64);

  // --- EdgeConv 4: x3 (C=128, pairs 64..127) -> pairs [128,256) ---
  k_sqnorm_pk<<<gN, 256, 0, stream>>>(xcatpk + 64L * N_PTS, bs256, 64, xx);
  k_knn<64><<<gKNN, 256, distLDS, stream>>>(xcatpk + 64L * N_PTS, bs256, xx, idx);
  k_edgeconv<128><<<dim3(N_PTS / 16, 256 / 16, BATCH), 32, 0, stream>>>(
      xcatpk + 64L * N_PTS, bs256, idx, w4, xcatpk, bs256, 128);

  // --- final: out = max_n lrelu(w5 @ xcat) -> [B, 1024] ---
  k_gemm_lrelu_maxn<512><<<dim3(1024 / 16, BATCH), 32, 0, stream>>>(
      xcatpk, bs256, w5, 1024, out);
}
